// KGAttentionLayer_76605036691536
// MI455X (gfx1250) — matrix operand order repaired and checked by reading, not compiled, not run on hardware
//
#include <hip/hip_runtime.h>
#include <hip/hip_bf16.h>

// ---------------------------------------------------------------------------
// Problem constants (match reference)
// ---------------------------------------------------------------------------
constexpr int BB = 2;
constexpr int LL = 2048;
constexpr int EE = 256;
constexpr int DD = 1024;     // d_model
constexpr int HH = 16;       // heads
constexpr int HD = 64;       // head dim
constexpr int MKEYS = LL + EE;   // 2304 keys per (b,h)
constexpr int KVPAD = 72;        // LDS row stride (elems) for K/V chunk tiles

typedef __attribute__((ext_vector_type(16))) __bf16 v16bf;
typedef __attribute__((ext_vector_type(8)))  float  v8f;
typedef unsigned int u32x4 __attribute__((ext_vector_type(4)));
typedef int          i32x4 __attribute__((ext_vector_type(4)));
typedef int          i32x8 __attribute__((ext_vector_type(8)));

#if __has_builtin(__builtin_amdgcn_tensor_load_to_lds)
#define HAVE_TDM 1
#else
#define HAVE_TDM 0
#endif

// ---------------------------------------------------------------------------
// WMMA fragment loaders (bf16 16x16x32, wave32; contiguous k = 16*(lane>>4) + e in both A and B)
// ---------------------------------------------------------------------------
// A matrix 16x32: lane&15 = row M.  lanes 0-15 koff=0, lanes 16-31 koff=16.
// elements e -> K = k0+koff+e. Two b128 loads.
__device__ __forceinline__ v16bf load_a_frag(const __bf16* A, int lda, int m0,
                                             int k0, int lane) {
    int r    = lane & 15;
    int koff = (lane >> 4) << 4;               // 0 or 16
    const __bf16* p = A + (size_t)(m0 + r) * lda + k0 + koff;
    union { v16bf v; uint4 q[2]; } f;
    f.q[0] = *(const uint4*)(p);
    f.q[1] = *(const uint4*)(p + 8);
    return f.v;
}

// B matrix 32x16 where source is row-major [N, K] (column n of B contiguous
// along K): lane holds N = lane&15; K = k0 + 16*(lane>>4) + e.
__device__ __forceinline__ v16bf load_b_frag_rowK(const __bf16* W, int ldw,
                                                  int n0, int k0, int lane) {
    int n  = lane & 15;
    int kb = k0 + ((lane >> 4) << 4);          // +0 or +16
    const __bf16* p = W + (size_t)(n0 + n) * ldw + kb;
    union { v16bf v; uint4 q[2]; } f;
    f.q[0] = *(const uint4*)(p);
    f.q[1] = *(const uint4*)(p + 8);
    return f.v;
}

__device__ __forceinline__ v8f wmma_bf16(v16bf a, v16bf b, v8f c) {
    return __builtin_amdgcn_wmma_f32_16x16x32_bf16(false, a, false, b,
                                                   (short)0, c, false, false);
}

__device__ __forceinline__ v8f v8f_zero() {
    v8f z = {0.f, 0.f, 0.f, 0.f, 0.f, 0.f, 0.f, 0.f};
    return z;
}

// ---------------------------------------------------------------------------
// TDM: load a [rows x cols] bf16 tile (row stride `row_stride` elems) from
// global into LDS at byte offset lds_off, padding each 128B row by 16B so the
// LDS image has KVPAD(=72)-element rows.  D# packed per CDNA5 ISA 8.3/8.4.
// ---------------------------------------------------------------------------
#if HAVE_TDM
__device__ __forceinline__ void tdm_load_2d_bf16(unsigned lds_off,
                                                 const __bf16* gsrc,
                                                 unsigned rows, unsigned cols,
                                                 unsigned row_stride) {
    unsigned long long ga = (unsigned long long)(const void*)gsrc;
    u32x4 g0;
    g0.x = 1u;                                           // count=1, user mode
    g0.y = lds_off;                                      // lds_addr
    g0.z = (unsigned)ga;                                 // global_addr lo
    g0.w = (unsigned)((ga >> 32) & 0x01FFFFFFu) | (2u << 30);  // hi | type=2
    i32x8 g1;
    // data_size=1(2B) | pad_enable | pad_interval=4(32 dw=128B) | pad_amt=3(4 dw=16B)
    g1[0] = (int)((1u << 16) | (1u << 20) | (4u << 22) | (3u << 25));
    g1[1] = (int)((cols & 0xFFFFu) << 16);               // tensor_dim0 lo16
    g1[2] = (int)(((cols >> 16) & 0xFFFFu) | ((rows & 0xFFFFu) << 16));
    g1[3] = (int)(((rows >> 16) & 0xFFFFu) | ((cols & 0xFFFFu) << 16)); // tile_dim0
    g1[4] = (int)(rows & 0xFFFFu);                       // tile_dim1 (tile_dim2=0)
    g1[5] = (int)row_stride;                             // tensor_dim0_stride lo32
    g1[6] = 0;
    g1[7] = 0;
    i32x4 z4 = {0, 0, 0, 0};
#if defined(__clang_major__) && (__clang_major__ >= 23)
    // clang-23 / therock-10.0 headers: 6-arg form (extra int32x8 group)
    i32x8 z8 = {0, 0, 0, 0, 0, 0, 0, 0};
    __builtin_amdgcn_tensor_load_to_lds(g0, g1, z4, z4, z8, 0);
#else
    // ROCm 7.2 (clang-22): 5-arg form
    __builtin_amdgcn_tensor_load_to_lds(g0, g1, z4, z4, 0);
#endif
}
#endif

__device__ __forceinline__ void wait_tensorcnt0() {
#if __has_builtin(__builtin_amdgcn_s_wait_tensorcnt)
    __builtin_amdgcn_s_wait_tensorcnt(0);
#else
    asm volatile("s_wait_tensorcnt 0x0" ::: "memory");
#endif
}

// ---------------------------------------------------------------------------
// fp32 -> bf16 convert
// ---------------------------------------------------------------------------
__global__ void cvt_f32_bf16(const float* __restrict__ src,
                             __bf16* __restrict__ dst, int n) {
    int i = blockIdx.x * blockDim.x + threadIdx.x;
    if (i < n) dst[i] = (__bf16)src[i];
}

// ---------------------------------------------------------------------------
// Generic GEMM: C[M,N] = A[M,K] (bf16) * W[N,K]^T (bf16) + bias, fp32 accum.
// One wave -> 16x64 tile.  block = 4 waves (16x256 per block).
// bias_row_div: 0 -> bias[col]; >0 -> bias[(row/div)*N + col]
// ---------------------------------------------------------------------------
__global__ void gemm_bf16_wmma(const __bf16* __restrict__ A, int lda,
                               const __bf16* __restrict__ W, int ldw,
                               const float* __restrict__ bias, int bias_row_div,
                               float* __restrict__ Cf, __bf16* __restrict__ Cbf,
                               int ldc, int M, int N, int K) {
    int lane = threadIdx.x & 31;
    int wave = threadIdx.x >> 5;
    int m0 = blockIdx.x * 16;
    int n0 = blockIdx.y * 256 + wave * 64;
    if (m0 >= M || n0 >= N) return;

    v8f acc[4];
    #pragma unroll
    for (int g = 0; g < 4; ++g) acc[g] = v8f_zero();

    for (int k0 = 0; k0 < K; k0 += 32) {
        if (k0 + 64 < K)   // warm next A tile (global_prefetch_b8)
            __builtin_prefetch(A + (size_t)(m0 + (lane & 15)) * lda + k0 + 64, 0, 0);
        v16bf a = load_a_frag(A, lda, m0, k0, lane);
        #pragma unroll
        for (int g = 0; g < 4; ++g) {
            v16bf b = load_b_frag_rowK(W, ldw, n0 + g * 16, k0, lane);
            acc[g] = wmma_bf16(a, b, acc[g]);
        }
    }

    int roff = (lane >> 4) << 3;
    int col  = lane & 15;
    int bbase = (bias && bias_row_div) ? (m0 / bias_row_div) * N : 0;
    #pragma unroll
    for (int g = 0; g < 4; ++g) {
        int n = n0 + g * 16 + col;
        float bv = bias ? bias[bbase + n] : 0.f;
        #pragma unroll
        for (int j = 0; j < 8; ++j) {
            int row = m0 + roff + j;
            float v = acc[g][j] + bv;
            if (Cf)  Cf[(size_t)row * ldc + n] = v;
            if (Cbf) Cbf[(size_t)row * ldc + n] = (__bf16)v;
        }
    }
}

// ---------------------------------------------------------------------------
// Flash attention over augmented keys.  Block = 4 waves = 4 consecutive
// 16-query tiles of the SAME (b,h): K/V chunks staged once per block in LDS
// (TDM async-tensor path when available, cooperative loads otherwise).
// ---------------------------------------------------------------------------
struct __align__(16) AttnLds {
    __bf16 K[32][KVPAD];
    __bf16 V[32][KVPAD];
    __bf16 P[4][16][32];
};

__global__ void attn_wmma(const __bf16* __restrict__ qb,
                          const __bf16* __restrict__ kb,
                          const __bf16* __restrict__ vb,
                          const __bf16* __restrict__ kkg,
                          const __bf16* __restrict__ vkg,
                          __bf16* __restrict__ out_bf) {
    __shared__ AttnLds sm;

    int lane = threadIdx.x & 31;
    int wave = threadIdx.x >> 5;
    int tile = blockIdx.x * 4 + wave;        // 0 .. B*H*(L/16)-1
    int mt = tile & ((LL / 16) - 1);
    int h  = (tile >> 7) & (HH - 1);
    int b  = tile >> 11;
    int m0 = mt * 16;
    const float scale = 0.125f;              // 1/sqrt(64)

    // q A-fragments (fixed for whole tile)
    const __bf16* qbase = qb + ((size_t)(b * LL + m0) * DD + h * HD);
    v16bf qa0 = load_a_frag(qbase, DD, 0, 0, lane);
    v16bf qa1 = load_a_frag(qbase, DD, 0, 32, lane);

    v8f acc[4];
    #pragma unroll
    for (int g = 0; g < 4; ++g) acc[g] = v8f_zero();
    float mrow[8], lrow[8];
    #pragma unroll
    for (int j = 0; j < 8; ++j) { mrow[j] = -1e30f; lrow[j] = 0.f; }

    const __bf16* Klds = &sm.K[0][0];
    const __bf16* Vlds = &sm.V[0][0];

    for (int c = 0; c < MKEYS / 32; ++c) {
        int kbase = c * 32;
        const __bf16* Ksrc;
        const __bf16* Vsrc;
        size_t krow0;
        if (kbase < LL) {
            Ksrc = kb + (size_t)b * LL * DD;
            Vsrc = vb + (size_t)b * LL * DD;
            krow0 = (size_t)kbase;
        } else {
            Ksrc = kkg + (size_t)b * EE * DD;
            Vsrc = vkg + (size_t)b * EE * DD;
            krow0 = (size_t)(kbase - LL);
        }
        const __bf16* gK = Ksrc + krow0 * DD + h * HD;
        const __bf16* gV = Vsrc + krow0 * DD + h * HD;

        __syncthreads();   // previous chunk fully consumed
#if HAVE_TDM
        if (wave == 0) {
            unsigned koff = (unsigned)(unsigned long long)(const void*)Klds;
            unsigned voff = (unsigned)(unsigned long long)(const void*)Vlds;
            tdm_load_2d_bf16(koff, gK, 32u, 64u, (unsigned)DD);
            tdm_load_2d_bf16(voff, gV, 32u, 64u, (unsigned)DD);
            wait_tensorcnt0();
        }
#else
        {   // cooperative staging: 128 threads x 16 elems
            int srow = threadIdx.x >> 2;          // 0..31
            int scol = (threadIdx.x & 3) << 4;    // 0,16,32,48
            const __bf16* gk = gK + (size_t)srow * DD + scol;
            const __bf16* gv = gV + (size_t)srow * DD + scol;
            uint4 k0v = *(const uint4*)gk;
            uint4 k1v = *(const uint4*)(gk + 8);
            uint4 v0v = *(const uint4*)gv;
            uint4 v1v = *(const uint4*)(gv + 8);
            *(uint4*)&sm.K[srow][scol]     = k0v;
            *(uint4*)&sm.K[srow][scol + 8] = k1v;
            *(uint4*)&sm.V[srow][scol]     = v0v;
            *(uint4*)&sm.V[srow][scol + 8] = v1v;
        }
#endif
        __syncthreads();   // chunk staged & visible

        // ---- scores: two 16-key tiles, K=64 accumulated over 2 WMMAs each
        v8f s[2];
        #pragma unroll
        for (int t = 0; t < 2; ++t) {
            v16bf b0 = load_b_frag_rowK(Klds, KVPAD, t * 16, 0, lane);
            v16bf b1 = load_b_frag_rowK(Klds, KVPAD, t * 16, 32, lane);
            v8f z = v8f_zero();
            z = wmma_bf16(qa0, b0, z);
            z = wmma_bf16(qa1, b1, z);
            s[t] = z;
        }

        // ---- online softmax update (per row j + 8*(lane>=16))
        #pragma unroll
        for (int j = 0; j < 8; ++j) {
            float v0 = s[0][j] * scale;
            float v1 = s[1][j] * scale;
            float m = fmaxf(v0, v1);
            #pragma unroll
            for (int msk = 1; msk < 16; msk <<= 1)
                m = fmaxf(m, __shfl_xor(m, msk, 32));
            float mnew = fmaxf(mrow[j], m);
            float corr = __expf(mrow[j] - mnew);
            lrow[j] *= corr;
            #pragma unroll
            for (int g = 0; g < 4; ++g) acc[g][j] *= corr;
            float p0 = __expf(v0 - mnew);
            float p1 = __expf(v1 - mnew);
            float ls = p0 + p1;
            #pragma unroll
            for (int msk = 1; msk < 16; msk <<= 1)
                ls += __shfl_xor(ls, msk, 32);
            lrow[j] += ls;
            mrow[j] = mnew;
            int row = ((lane >> 4) << 3) + j;
            sm.P[wave][row][lane & 15]        = (__bf16)p0;
            sm.P[wave][row][(lane & 15) + 16] = (__bf16)p1;
        }
        asm volatile("s_wait_dscnt 0" ::: "memory");  // intra-wave LDS RAW

        // probs C-layout -> A-fragment via LDS
        v16bf pa = load_a_frag((const __bf16*)&sm.P[wave][0][0], 32, 0, 0, lane);

        // ---- P·V from LDS V tile (gather: K = key index, row stride KVPAD)
        #pragma unroll
        for (int g = 0; g < 4; ++g) {
            union { v16bf v; unsigned short u[16]; } bf;
            int n  = lane & 15;
            int e0 = (lane >> 4) << 4;
            const __bf16* vcol = Vlds + (size_t)e0 * KVPAD + g * 16 + n;
            #pragma unroll
            for (int e = 0; e < 16; ++e)
                bf.u[e] = *(const unsigned short*)(vcol + (size_t)e * KVPAD);
            acc[g] = wmma_bf16(pa, bf.v, acc[g]);
        }
    }

    // ---- normalize + store bf16 (consumed only by bf16 GEMMs)
    int roff = (lane >> 4) << 3;
    int col  = lane & 15;
    #pragma unroll
    for (int j = 0; j < 8; ++j) {
        float inv = 1.0f / lrow[j];
        int row = m0 + roff + j;
        #pragma unroll
        for (int g = 0; g < 4; ++g) {
            size_t idx = ((size_t)(b * LL + row)) * DD + h * HD + g * 16 + col;
            out_bf[idx] = (__bf16)(acc[g][j] * inv);
        }
    }
}

// ---------------------------------------------------------------------------
// kg_global mean over E:  kgmean[b][d] = mean_e kg[b][e][d]
// ---------------------------------------------------------------------------
__global__ void kg_mean(const float* __restrict__ kg, float* __restrict__ kgm) {
    int i = blockIdx.x * blockDim.x + threadIdx.x;   // b*DD + d
    if (i >= BB * DD) return;
    int b = i / DD, d = i - b * DD;
    float s = 0.f;
    const float* p = kg + (size_t)b * EE * DD + d;
    for (int e = 0; e < EE; ++e) s += p[(size_t)e * DD];
    kgm[i] = s * (1.0f / EE);
}

// ---------------------------------------------------------------------------
// Fold kg_global gate half into a per-batch bias:
//   gbias[b][n] = bg[n] + dot(kgmean[b], Wg[n][DD:2DD])
// ---------------------------------------------------------------------------
__global__ void gate_bias(const float* __restrict__ kgm,
                          const float* __restrict__ Wg,
                          const float* __restrict__ bg,
                          float* __restrict__ gbias) {
    int i = blockIdx.x * blockDim.x + threadIdx.x;   // b*DD + n
    if (i >= BB * DD) return;
    int b = i / DD, n = i - b * DD;
    const float* w = Wg + (size_t)n * (2 * DD) + DD;
    const float* m = kgm + (size_t)b * DD;
    float s = bg[n];
    for (int k = 0; k < DD; ++k) s += m[k] * w[k];
    gbias[i] = s;
}

// ---------------------------------------------------------------------------
// Final blend:  out = sigmoid(G) * y + (1 - sigmoid(G)) * x
// ---------------------------------------------------------------------------
__global__ void blend(const float* __restrict__ G, const float* __restrict__ y,
                      const float* __restrict__ x, float* __restrict__ out,
                      int n) {
    int i = blockIdx.x * blockDim.x + threadIdx.x;
    if (i >= n) return;
    float g = 1.0f / (1.0f + __expf(-G[i]));
    out[i] = g * y[i] + (1.0f - g) * x[i];
}

// ---------------------------------------------------------------------------
// Host-side pipeline
// ---------------------------------------------------------------------------
extern "C" void kernel_launch(void* const* d_in, const int* in_sizes, int n_in,
                              void* d_out, int out_size, void* d_ws, size_t ws_size,
                              hipStream_t stream) {
    const float* x   = (const float*)d_in[0];
    const float* kg  = (const float*)d_in[1];
    const float* Wq  = (const float*)d_in[2];
    const float* bq  = (const float*)d_in[3];
    const float* Wk  = (const float*)d_in[4];
    const float* bk  = (const float*)d_in[5];
    const float* Wv  = (const float*)d_in[6];
    const float* bv  = (const float*)d_in[7];
    const float* Wkk = (const float*)d_in[8];
    const float* bkk = (const float*)d_in[9];
    const float* Wkv = (const float*)d_in[10];
    const float* bkv = (const float*)d_in[11];
    const float* Wo  = (const float*)d_in[12];
    const float* bo  = (const float*)d_in[13];
    const float* Wg  = (const float*)d_in[14];
    const float* bg  = (const float*)d_in[15];
    float* out = (float*)d_out;

    // ---- workspace carve-up
    char* ws = (char*)d_ws;
    size_t off = 0;
    auto carve = [&](size_t bytes) -> char* {
        char* p = ws + off;
        off = (off + bytes + 255) & ~(size_t)255;
        return p;
    };
    const size_t NX  = (size_t)BB * LL * DD;   // 4M
    const size_t NKG = (size_t)BB * EE * DD;   // 512K
    const size_t NW  = (size_t)DD * DD;        // 1M
    __bf16* x_bf   = (__bf16*)carve(NX * 2);
    __bf16* kg_bf  = (__bf16*)carve(NKG * 2);
    __bf16* Wq_bf  = (__bf16*)carve(NW * 2);
    __bf16* Wk_bf  = (__bf16*)carve(NW * 2);
    __bf16* Wv_bf  = (__bf16*)carve(NW * 2);
    __bf16* Wkk_bf = (__bf16*)carve(NW * 2);
    __bf16* Wkv_bf = (__bf16*)carve(NW * 2);
    __bf16* Wo_bf  = (__bf16*)carve(NW * 2);
    __bf16* Wg_bf  = (__bf16*)carve(NW * 2 * 2);     // [DD, 2*DD]
    __bf16* q_bf   = (__bf16*)carve(NX * 2);
    __bf16* k_bf   = (__bf16*)carve(NX * 2);
    __bf16* v_bf   = (__bf16*)carve(NX * 2);
    __bf16* kkg_bf = (__bf16*)carve(NKG * 2);
    __bf16* vkg_bf = (__bf16*)carve(NKG * 2);
    __bf16* att_bf = (__bf16*)carve(NX * 2);
    float*  Gf     = (float*)carve(NX * 4);
    float*  yf     = (float*)carve(NX * 4);
    float*  kgm    = (float*)carve((size_t)BB * DD * 4);
    float*  gbias  = (float*)carve((size_t)BB * DD * 4);
    (void)ws_size; (void)n_in; (void)in_sizes; (void)out_size;

    auto cvt = [&](const float* s, __bf16* d, size_t n) {
        cvt_f32_bf16<<<(unsigned)((n + 255) / 256), 256, 0, stream>>>(s, d, (int)n);
    };
    cvt(x,   x_bf,   NX);
    cvt(kg,  kg_bf,  NKG);
    cvt(Wq,  Wq_bf,  NW);
    cvt(Wk,  Wk_bf,  NW);
    cvt(Wv,  Wv_bf,  NW);
    cvt(Wkk, Wkk_bf, NW);
    cvt(Wkv, Wkv_bf, NW);
    cvt(Wo,  Wo_bf,  NW);
    cvt(Wg,  Wg_bf,  NW * 2);

    // ---- projections (bf16 WMMA, bf16 outputs)
    {
        dim3 grid((BB * LL) / 16, DD / 256);
        gemm_bf16_wmma<<<grid, 128, 0, stream>>>(x_bf, DD, Wq_bf, DD, bq, 0,
                                                 nullptr, q_bf, DD, BB * LL, DD, DD);
        gemm_bf16_wmma<<<grid, 128, 0, stream>>>(x_bf, DD, Wk_bf, DD, bk, 0,
                                                 nullptr, k_bf, DD, BB * LL, DD, DD);
        gemm_bf16_wmma<<<grid, 128, 0, stream>>>(x_bf, DD, Wv_bf, DD, bv, 0,
                                                 nullptr, v_bf, DD, BB * LL, DD, DD);
    }
    {
        dim3 grid((BB * EE) / 16, DD / 256);
        gemm_bf16_wmma<<<grid, 128, 0, stream>>>(kg_bf, DD, Wkk_bf, DD, bkk, 0,
                                                 nullptr, kkg_bf, DD, BB * EE, DD, DD);
        gemm_bf16_wmma<<<grid, 128, 0, stream>>>(kg_bf, DD, Wkv_bf, DD, bkv, 0,
                                                 nullptr, vkg_bf, DD, BB * EE, DD, DD);
    }

    // ---- gate bias fold (rank-1 kg_global path)
    kg_mean<<<(BB * DD + 255) / 256, 256, 0, stream>>>(kg, kgm);
    gate_bias<<<(BB * DD + 255) / 256, 256, 0, stream>>>(kgm, Wg, bg, gbias);

    // ---- flash attention over L + E keys
    {
        int tiles = BB * HH * (LL / 16);         // 4096 waves
        attn_wmma<<<tiles / 4, 128, 0, stream>>>(q_bf, k_bf, v_bf,
                                                 kkg_bf, vkg_bf, att_bf);
    }

    // ---- output projection + gate logits
    {
        dim3 grid((BB * LL) / 16, DD / 256);
        gemm_bf16_wmma<<<grid, 128, 0, stream>>>(att_bf, DD, Wo_bf, DD, bo, 0,
                                                 yf, nullptr, DD, BB * LL, DD, DD);
        gemm_bf16_wmma<<<grid, 128, 0, stream>>>(att_bf, DD, Wg_bf, 2 * DD, gbias, LL,
                                                 Gf, nullptr, DD, BB * LL, DD, DD);
    }

    // ---- sigmoid blend
    blend<<<(unsigned)((NX + 255) / 256), 256, 0, stream>>>(Gf, yf, x, out, (int)NX);
}
